// MySelfAttention_76922864271592
// MI455X (gfx1250) — matrix operand order repaired and checked by reading, not compiled, not run on hardware
//
#include <hip/hip_runtime.h>
#include <hip/hip_bf16.h>
#include <math.h>

typedef __attribute__((ext_vector_type(16))) _Float16 v16h;
typedef __attribute__((ext_vector_type(8)))  _Float16 v8h;
typedef __attribute__((ext_vector_type(8)))  float    v8f;

#define CC   768
#define HH   12
#define TT   1024
#define BB   8
#define HD   64
#define N3C  2304
#define MTOT 8192   // B*T

// ---------------------------------------------------------------------------
// WMMA fragment loaders (16x16x32 f16; k order is this code's contiguous choice, same for A and B)
// A (16x32, MxK): lane m = lane&15, g = lane>>4.
//   halves 0-15 -> K = 16*g .. 16*g+15     (contiguous)
// B (32x16, KxN): lane n = lane&15, g = lane>>4; halves 0-15 -> K = 16*g + h.
//   All B operands are stored "k-contiguous" ([n][k]) -> one 32B load/lane.
// C/D (16x16 f32): lane n = lane&15; VGPR r -> M = r + 8*(lane>>4).
// ---------------------------------------------------------------------------
__device__ __forceinline__ v16h load_a_frag(const _Float16* __restrict__ base, int ld) {
    int lane = threadIdx.x & 31;
    const _Float16* p = base + (lane & 15) * ld + ((lane >> 4) << 4);
    v8h lo = *(const v8h*)(p);
    v8h hi = *(const v8h*)(p + 8);
    v16h a;
#pragma unroll
    for (int i = 0; i < 8; ++i) { a[i] = lo[i]; a[i + 8] = hi[i]; }
    return a;
}

__device__ __forceinline__ v16h load_b_frag(const _Float16* __restrict__ base, int ld) {
    int lane = threadIdx.x & 31;
    const _Float16* p = base + (lane & 15) * ld + ((lane >> 4) << 4);
    return *(const v16h*)(p);
}

__device__ __forceinline__ v8f wmma_f16(v16h a, v16h b, v8f c) {
    return __builtin_amdgcn_wmma_f32_16x16x32_f16(false, a, false, b, (short)0, c,
                                                  false, false);
}

// CDNA5 async copy global->LDS (ASYNCcnt path). LDS address = low 32 bits of
// the generic pointer (flat aperture maps addr[31:0] to the LDS offset).
__device__ __forceinline__ void async_ld_b128(uint32_t lds_addr,
                                              const _Float16* gptr) {
    asm volatile("global_load_async_to_lds_b128 %0, %1, off"
                 :: "v"(lds_addr), "v"(gptr)
                 : "memory");
}
__device__ __forceinline__ void wait_async0() {
    asm volatile("s_wait_asynccnt 0x0" ::: "memory");
}

// ---------------------------------------------------------------------------
// Converters
// ---------------------------------------------------------------------------
__global__ void f32_to_f16_kernel(const float* __restrict__ in,
                                  _Float16* __restrict__ out, int n) {
    int i = blockIdx.x * blockDim.x + threadIdx.x;
    if (i < n) out[i] = (_Float16)in[i];
}

// in is [K][N] f32 (y = x @ W); out is [N][K] f16 (k-contiguous for B frags)
__global__ void transpose_f32_to_f16_kernel(const float* __restrict__ in,
                                            _Float16* __restrict__ out,
                                            int K, int N) {
    int i = blockIdx.x * blockDim.x + threadIdx.x;
    if (i < K * N) {
        int k = i / N, n = i - k * N;
        out[n * K + k] = (_Float16)in[i];
    }
}

// ---------------------------------------------------------------------------
// Wave-tile GEMM core: block = 8 waves sharing one 64-col tile, each wave a
// 16-row tile. B tile (64n x 32k = 4KB) staged per k-step into LDS by async
// copies (256 lanes x 16B), double-buffered so the copy for step i+1 overlaps
// the WMMAs of step i.
// ---------------------------------------------------------------------------
struct GemmAcc { v8f a[4]; };

template <typename EPI>
__device__ __forceinline__ void gemm_16x64(const _Float16* __restrict__ A,
                                           const _Float16* __restrict__ Wt,
                                           int m0, int n0,
                                           _Float16 (*Bst)[64 * 32],
                                           EPI&& epilogue) {
    int lane  = threadIdx.x & 31;
    int col16 = lane & 15, g = lane >> 4;
    int t     = threadIdx.x;

    // thread t copies 16B: row n=t>>2 of the tile, k-chunk (t&3)*8
    const _Float16* gsrc0 = Wt + (size_t)(n0 + (t >> 2)) * CC + ((t & 3) << 3);

    async_ld_b128((uint32_t)(uintptr_t)&Bst[0][t * 8], gsrc0);
    wait_async0();
    __syncthreads();

    v8f acc[4] = {};
    for (int i = 0; i < 24; ++i) {
        if (i + 1 < 24)
            async_ld_b128((uint32_t)(uintptr_t)&Bst[(i + 1) & 1][t * 8],
                          gsrc0 + (i + 1) * 32);
        if (i + 1 < 24)
            __builtin_prefetch(A + m0 * CC + (i + 1) * 32, 0, 3);
        v16h a = load_a_frag(A + m0 * CC + i * 32, CC);
        const _Float16* bb = &Bst[i & 1][0];
#pragma unroll
        for (int s = 0; s < 4; ++s) {
            v16h b = *(const v16h*)(bb + (16 * s + col16) * 32 + (g << 4));
            acc[s] = wmma_f16(a, b, acc[s]);
        }
        wait_async0();      // copy for next buffer complete (this wave)
        __syncthreads();    // all waves copied; all waves done reading buf i&1
    }
    epilogue(acc, col16, g);
}

// ---------------------------------------------------------------------------
// QKV GEMM:  [8192,768]f16 x [768,2304] (stored [2304,768] f16) + bias.
// Epilogue scatters into Qh/Kh [B,H,T,64] (Q scaled by 0.125), Vt [B,H,64,T].
// ---------------------------------------------------------------------------
__global__ __launch_bounds__(256) void qkv_gemm_kernel(
    const _Float16* __restrict__ Xh, const _Float16* __restrict__ Wt,
    const float* __restrict__ bias, _Float16* __restrict__ Qh,
    _Float16* __restrict__ Kh, _Float16* __restrict__ Vt) {
    __shared__ _Float16 Bst[2][64 * 32];
    int wid = blockIdx.x * 8 + (threadIdx.x >> 5);
    int m0 = (wid & 511) << 4;   // 512 row tiles
    int n0 = (wid >> 9) << 6;    // 36 col tiles of 64 (uniform in block)

    gemm_16x64(Xh, Wt, m0, n0, Bst,
        [&](v8f (&acc)[4], int col16, int g) {
#pragma unroll
            for (int s = 0; s < 4; ++s) {
                int n = n0 + 16 * s + col16;
                float bv = bias[n];
                int which = n / CC;      // 0=q 1=k 2=v
                int cc = n - which * CC;
                int h = cc >> 6, hd = cc & 63;
#pragma unroll
                for (int r = 0; r < 8; ++r) {
                    int m = m0 + r + 8 * g;
                    int b_ = m >> 10, tq = m & 1023;
                    float val = acc[s][r] + bv;
                    if (which == 0)
                        Qh[((b_ * HH + h) * TT + tq) * HD + hd] =
                            (_Float16)(val * 0.125f);
                    else if (which == 1)
                        Kh[((b_ * HH + h) * TT + tq) * HD + hd] = (_Float16)val;
                    else
                        Vt[((b_ * HH + h) * HD + hd) * TT + tq] = (_Float16)val;
                }
            }
        });
}

// ---------------------------------------------------------------------------
// Output projection: [8192,768]f16 x [768,768]^T f16 -> f32 + bias
// ---------------------------------------------------------------------------
__global__ __launch_bounds__(256) void proj_gemm_kernel(
    const _Float16* __restrict__ Yh, const _Float16* __restrict__ Wt,
    const float* __restrict__ bias, float* __restrict__ out) {
    __shared__ _Float16 Bst[2][64 * 32];
    int wid = blockIdx.x * 8 + (threadIdx.x >> 5);
    int m0 = (wid & 511) << 4;
    int n0 = (wid >> 9) << 6;            // 12 col tiles of 64

    gemm_16x64(Yh, Wt, m0, n0, Bst,
        [&](v8f (&acc)[4], int col16, int g) {
#pragma unroll
            for (int s = 0; s < 4; ++s) {
                int n = n0 + 16 * s + col16;
                float bv = bias[n];
#pragma unroll
                for (int r = 0; r < 8; ++r) {
                    int m = m0 + r + 8 * g;
                    out[(size_t)m * CC + n] = acc[s][r] + bv;
                }
            }
        });
}

// ---------------------------------------------------------------------------
// Attention: one block per (b, h, 16-query tile). 256 threads = 8 waves.
// ---------------------------------------------------------------------------
__global__ __launch_bounds__(256) void attn_kernel(
    const _Float16* __restrict__ Qh, const _Float16* __restrict__ Kh,
    const _Float16* __restrict__ Vt, const int* __restrict__ mask,
    _Float16* __restrict__ Yh) {
    __shared__ _Float16 Sbuf[16][TT];      // scores, then P (in place)
    __shared__ float    wmax[8][16];
    __shared__ float    rowmax[16];
    __shared__ float    rinv[16];
    __shared__ float    obuf[4][16][16];

    int bid  = blockIdx.x;
    int qblk = bid & 63;
    int h    = (bid >> 6) % HH;
    int b    = bid / (64 * HH);
    int q0   = qblk << 4;

    int w     = threadIdx.x >> 5;
    int lane  = threadIdx.x & 31;
    int col16 = lane & 15, g = lane >> 4;

    const float NEGINF = -__builtin_inff();

    // ---- Phase 1: S = (Q*0.125) @ K^T, causal mask, per-wave row maxes ----
    const _Float16* Qbase = Qh + ((b * HH + h) * TT + q0) * HD;
    v16h aq0 = load_a_frag(Qbase + 0, HD);
    v16h aq1 = load_a_frag(Qbase + 32, HD);

    float lmax[8];
#pragma unroll
    for (int r = 0; r < 8; ++r) lmax[r] = NEGINF;

    for (int i = 0; i < 8; ++i) {
        int j = w + 8 * i;                 // key tile (wave-uniform)
        v8f c = {};
        if (j <= qblk) {                   // tile has at least one visible key
            const _Float16* Kbase = Kh + ((b * HH + h) * TT + (j << 4)) * HD;
            v16h bk0 = load_b_frag(Kbase + 0, HD);
            v16h bk1 = load_b_frag(Kbase + 32, HD);
            c = wmma_f16(aq0, bk0, c);
            c = wmma_f16(aq1, bk1, c);
        }
        int keyn = (j << 4) + col16;
#pragma unroll
        for (int r = 0; r < 8; ++r) {
            int qrow = r + 8 * g;
            float s = (keyn <= q0 + qrow) ? c[r] : NEGINF;
            Sbuf[qrow][(j << 4) + col16] = (_Float16)s;
            lmax[r] = fmaxf(lmax[r], s);
        }
    }
#pragma unroll
    for (int r = 0; r < 8; ++r) {
        float v = lmax[r];
#pragma unroll
        for (int d = 1; d < 16; d <<= 1) v = fmaxf(v, __shfl_xor(v, d, 32));
        if (col16 == 0) wmax[w][8 * g + r] = v;
    }
    __syncthreads();

    // ---- Phase 2: softmax (handles mask==0 -> uniform row exactly) ----
    if (threadIdx.x < 16) {
        float m = NEGINF;
#pragma unroll
        for (int ww = 0; ww < 8; ++ww) m = fmaxf(m, wmax[ww][threadIdx.x]);
        rowmax[threadIdx.x] = m;
    }
    __syncthreads();

    {
        int row = threadIdx.x >> 4;        // 16 threads per row (one half-wave)
        int cs  = threadIdx.x & 15;
        bool rmask = (mask[b * TT + q0 + row] == 0);
        float mx = rowmax[row];
        float psum = 0.f;
        for (int i = 0; i < 64; ++i) {
            int col = cs + 16 * i;
            float e = rmask ? 1.0f : expf((float)Sbuf[row][col] - mx);
            psum += e;
            Sbuf[row][col] = (_Float16)e;  // unnormalized P, in place
        }
#pragma unroll
        for (int d = 1; d < 16; d <<= 1) psum += __shfl_xor(psum, d, 32);
        if (cs == 0) rinv[row] = 1.0f / psum;
    }
    __syncthreads();

    // ---- Phase 3: O = P @ V; waves split key dim in half, combine in LDS ----
    int hdt   = w & 3;                     // HD tile (16 cols)
    int khalf = w >> 2;                    // key half
    const _Float16* Vbase = Vt + ((b * HH + h) * HD + hdt * 16) * TT;
    v8f o = {};
    for (int i = 0; i < 16; ++i) {
        int k0 = khalf * 512 + i * 32;
        v16h ap = load_a_frag(&Sbuf[0][0] + k0, TT);
        v16h bv = load_b_frag(Vbase + k0, TT);
        o = wmma_f16(ap, bv, o);
    }
    if (w < 4) {
#pragma unroll
        for (int r = 0; r < 8; ++r) obuf[w][r + 8 * g][col16] = o[r];
    }
    __syncthreads();
    if (w >= 4) {
        int t = w - 4;                     // matches hdt of partner wave
#pragma unroll
        for (int r = 0; r < 8; ++r) {
            int qrow = r + 8 * g;
            float val = (o[r] + obuf[t][qrow][col16]) * rinv[qrow];
            Yh[((size_t)(b * TT + q0 + qrow)) * CC + h * HD + t * 16 + col16] =
                (_Float16)val;
        }
    }
}

// ---------------------------------------------------------------------------
extern "C" void kernel_launch(void* const* d_in, const int* in_sizes, int n_in,
                              void* d_out, int out_size, void* d_ws, size_t ws_size,
                              hipStream_t stream) {
    const float* x      = (const float*)d_in[0];
    const int*   mask   = (const int*)d_in[1];
    const float* W_attn = (const float*)d_in[2];
    const float* b_attn = (const float*)d_in[3];
    const float* W_proj = (const float*)d_in[4];
    const float* b_proj = (const float*)d_in[5];
    float* out = (float*)d_out;

    char* ws = (char*)d_ws;
    _Float16* Xh  = (_Float16*)ws;  ws += (size_t)MTOT * CC * 2;   // 12.6 MB
    _Float16* Wat = (_Float16*)ws;  ws += (size_t)N3C * CC * 2;    //  3.5 MB
    _Float16* Wpt = (_Float16*)ws;  ws += (size_t)CC * CC * 2;     //  1.2 MB
    _Float16* Qh  = (_Float16*)ws;  ws += (size_t)MTOT * CC * 2;
    _Float16* Kh  = (_Float16*)ws;  ws += (size_t)MTOT * CC * 2;
    _Float16* Vt  = (_Float16*)ws;  ws += (size_t)MTOT * CC * 2;
    _Float16* Yh  = (_Float16*)ws;  ws += (size_t)MTOT * CC * 2;   // ~68 MB total

    f32_to_f16_kernel<<<(MTOT * CC) / 256, 256, 0, stream>>>(x, Xh, MTOT * CC);
    transpose_f32_to_f16_kernel<<<(CC * N3C) / 256, 256, 0, stream>>>(W_attn, Wat, CC, N3C);
    transpose_f32_to_f16_kernel<<<(CC * CC) / 256, 256, 0, stream>>>(W_proj, Wpt, CC, CC);

    // 512 m-tiles * 36 n-tiles = 18432 waves / 8 per block
    qkv_gemm_kernel<<<2304, 256, 0, stream>>>(Xh, Wat, b_attn, Qh, Kh, Vt);

    // one block per (b, h, 16-query tile)
    attn_kernel<<<BB * HH * (TT / 16), 256, 0, stream>>>(Qh, Kh, Vt, mask, Yh);

    // 512 m-tiles * 12 n-tiles = 6144 waves / 8 per block
    proj_gemm_kernel<<<768, 256, 0, stream>>>(Yh, Wpt, b_proj, out);
}
